// BINDC_87668872446064
// MI455X (gfx1250) — matrix operand order repaired and checked by reading, not compiled, not run on hardware
//
#include <hip/hip_runtime.h>
#include <hip/hip_bf16.h>
#include <cstdint>

typedef __bf16 bf16_t;
typedef __attribute__((ext_vector_type(16))) __bf16 v16bf;
typedef __attribute__((ext_vector_type(8)))  float  v8f;
typedef __attribute__((ext_vector_type(4)))  int    v4i;

#define B_   4
#define S_   1024
#define H_   1024
#define L_   4
#define NH_  16
#define NKV_ 8
#define HD_  64
#define F_   3072
#define W_   12
#define EPS_ 1e-6f
#define NEG_ -1e9f

// gfx1250 async global->LDS copy (ASYNCcnt-tracked), guarded so both toolchains compile.
#define HAS_ASYNC (__has_builtin(__builtin_amdgcn_global_load_async_to_lds_b128) && \
                   __has_builtin(__builtin_amdgcn_s_wait_asynccnt))

#if HAS_ASYNC
typedef __attribute__((address_space(3))) v4i lds_v4i;
__device__ __forceinline__ void async_cp16(const bf16_t* g, bf16_t* l) {
  // signature (per hipcc diagnostic): param0 = generic v4i* (global addr),
  // param1 = LDS-side pointer, then imm offset, imm cpol.
  __builtin_amdgcn_global_load_async_to_lds_b128((v4i*)g, (lds_v4i*)l, 0, 0);
}
#endif

// ---------------------------------------------------------------- helpers
__device__ __forceinline__ v8f v8f_zero() {
  v8f v;
#pragma unroll
  for (int i = 0; i < 8; ++i) v[i] = 0.0f;
  return v;
}

// A-fragment (16x32 bf16, row-major source, `stride` elements per row).
// Lanes 0-15: row M=lane, K {0..7}(v0..3) + {16..23}(v4..7); lanes 16-31: K {8..15}+{24..31}.
__device__ __forceinline__ v16bf load_a_frag(const bf16_t* p0, int lane, int stride) {
  int row = lane & 15;
  int hi  = (lane >> 4) << 3;
  const bf16_t* p = p0 + row * stride + hi;
  v16bf f;
  ((v4i*)&f)[0] = *(const v4i*)(p);
  ((v4i*)&f)[1] = *(const v4i*)(p + 16);
  return f;
}

// B-fragment (32x16) from Bs[n][k] layout: lanes 0-15 col N=lane K {0..7}+{16..23}; lanes 16-31 K {8..15}+{24..31}.
__device__ __forceinline__ v16bf load_b_frag(const bf16_t* p0, int lane, int stride) {
  int col = lane & 15;
  int hi  = (lane >> 4) << 3;
  const bf16_t* p = p0 + col * stride + hi;
  v16bf f;
  ((v4i*)&f)[0] = *(const v4i*)(p);
  ((v4i*)&f)[1] = *(const v4i*)(p + 16);
  return f;
}

// ---------------------------------------------------------------- small kernels
// Transpose-convert one weight: in [L?][K][N] f32 -> out [L?][N][K] bf16 (layer via blockIdx.z)
__global__ void cvt_transpose(const float* __restrict__ in, bf16_t* __restrict__ out,
                              int K, int N) {
  __shared__ float tile[32][33];
  long lofs = (long)blockIdx.z * K * N;
  int n0 = blockIdx.x * 32, k0 = blockIdx.y * 32;
  int tx = threadIdx.x, ty = threadIdx.y;   // (32, 8)
#pragma unroll
  for (int i = 0; i < 4; ++i)
    tile[ty + i * 8][tx] = in[lofs + (long)(k0 + ty + i * 8) * N + n0 + tx];
  __syncthreads();
#pragma unroll
  for (int i = 0; i < 4; ++i)
    out[lofs + (long)(n0 + ty + i * 8) * K + k0 + tx] = (bf16_t)tile[tx][ty + i * 8];
}

__global__ void embed_gather(const int* __restrict__ ids, const float* __restrict__ emb,
                             float* __restrict__ h) {
  int t = blockIdx.x;
  long id = ids[t];
  const float4* s = (const float4*)(emb + id * (long)H_);
  float4* d = (float4*)(h + (long)t * H_);
  d[threadIdx.x] = s[threadIdx.x];   // 256 threads * 4 = H_
}

__global__ void rmsnorm_kernel(const float* __restrict__ h, const float* __restrict__ w,
                               bf16_t* __restrict__ out_bf, float* __restrict__ out_f) {
  int row = blockIdx.x;
  int tid = threadIdx.x;
  const float4* src = (const float4*)(h + (long)row * H_);
  float4 v = src[tid];
  float ss = v.x * v.x + v.y * v.y + v.z * v.z + v.w * v.w;
#pragma unroll
  for (int m = 16; m; m >>= 1) ss += __shfl_xor(ss, m, 32);
  __shared__ float red[8];
  int wave = tid >> 5, lane = tid & 31;
  if (lane == 0) red[wave] = ss;
  __syncthreads();
  float tot = 0.f;
#pragma unroll
  for (int i = 0; i < 8; ++i) tot += red[i];
  float inv = rsqrtf(tot * (1.0f / H_) + EPS_);
  float4 wv = ((const float4*)w)[tid];
  float o0 = v.x * inv * wv.x, o1 = v.y * inv * wv.y;
  float o2 = v.z * inv * wv.z, o3 = v.w * inv * wv.w;
  if (out_bf) {
    bf16_t* ob = out_bf + (long)row * H_ + tid * 4;
    ob[0] = (bf16_t)o0; ob[1] = (bf16_t)o1; ob[2] = (bf16_t)o2; ob[3] = (bf16_t)o3;
  }
  if (out_f) {
    ((float4*)(out_f + (long)row * H_))[tid] = make_float4(o0, o1, o2, o3);
  }
}

// Per (token, head): RMS over HD then RoPE; output head-major bf16 [B, nheads, S, HD].
__global__ void qk_post(const float* __restrict__ qin, const float* __restrict__ nw,
                        bf16_t* __restrict__ out, int nheads) {
  int s = blockIdx.x, hh = blockIdx.y, b = blockIdx.z;
  int lane = threadIdx.x;   // 32
  long base = (((long)(b * S_ + s)) * nheads + hh) * HD_;
  float x0 = qin[base + lane];
  float x1 = qin[base + lane + 32];
  float ss = x0 * x0 + x1 * x1;
#pragma unroll
  for (int m = 16; m; m >>= 1) ss += __shfl_xor(ss, m, 32);
  float inv = rsqrtf(ss * (1.0f / HD_) + EPS_);
  x0 = x0 * inv * nw[lane];
  x1 = x1 * inv * nw[lane + 32];
  float f = (float)s * __powf(1000000.0f, -(float)lane * (1.0f / 32.0f));
  float c = __cosf(f), sn = __sinf(f);
  float o0 = x0 * c - x1 * sn;
  float o1 = x1 * c + x0 * sn;
  long ob = (((long)(b * nheads + hh)) * S_ + s) * HD_;
  out[ob + lane]      = (bf16_t)o0;
  out[ob + lane + 32] = (bf16_t)o1;
}

// [B,S,NKV,HD] fp32 -> [B,NKV,HD,S] bf16 (d-major so attention V tile is a direct copy)
__global__ void v_pack(const float* __restrict__ vin, bf16_t* __restrict__ out) {
  long n = (long)B_ * S_ * NKV_ * HD_;
  long i = (long)blockIdx.x * blockDim.x + threadIdx.x;
  if (i >= n) return;
  int d = i & (HD_ - 1);
  long t = i >> 6;
  int kv = t & (NKV_ - 1);
  long bs = t >> 3;
  int s = bs & (S_ - 1);
  int b = (int)(bs >> 10);
  out[(((long)(b * NKV_ + kv)) * HD_ + d) * S_ + s] = (bf16_t)vin[i];
}

__global__ void silu_mul(const float* __restrict__ g, const float* __restrict__ u,
                         bf16_t* __restrict__ out, long n) {
  long i = (long)blockIdx.x * blockDim.x + threadIdx.x;
  if (i >= n) return;
  float x = g[i];
  float s = x * (1.0f / (1.0f + __expf(-x)));
  out[i] = (bf16_t)(s * u[i]);
}

__global__ void add_res(float* __restrict__ h, const float* __restrict__ c, long n) {
  long i = (long)blockIdx.x * blockDim.x + threadIdx.x;
  if (i < n) h[i] += c[i];
}

// ---------------------------------------------------------------- GEMM (bf16 WMMA)
// C[M,N] = A[M,K] * Bt[N,K]^T, double-buffered LDS, async or reg-prefetch staging.
#define BM 128
#define BN 128
#define BK 32
#define LDT 40   // BK + 8 pad (80B rows, 16B aligned)

__global__ __launch_bounds__(256) void gemm_bf16(const bf16_t* __restrict__ A,
                                                 const bf16_t* __restrict__ Bt,
                                                 float* __restrict__ C,
                                                 int M, int N, int K) {
  __shared__ bf16_t As[2 * BM * LDT];
  __shared__ bf16_t Bs[2 * BN * LDT];
  int tid = threadIdx.x;
  int bn = blockIdx.x * BN, bm = blockIdx.y * BM;
  int wave = tid >> 5, lane = tid & 31;
  int wm = (wave >> 1) * 32, wn = (wave & 1) * 64;

  int rc[2], sc[2];
  const bf16_t* gA[2];
  const bf16_t* gB[2];
#pragma unroll
  for (int i = 0; i < 2; ++i) {
    int c = tid + i * 256;           // 512 16B chunks per tile
    rc[i] = c >> 2; sc[i] = (c & 3) * 8;
    gA[i] = A  + (long)(bm + rc[i]) * K + sc[i];
    gB[i] = Bt + (long)(bn + rc[i]) * K + sc[i];
  }
  v8f acc[2][4];
#pragma unroll
  for (int i = 0; i < 2; ++i)
#pragma unroll
    for (int j = 0; j < 4; ++j) acc[i][j] = v8f_zero();

  // prologue: stage k-step 0 into buffer 0
#if HAS_ASYNC
#pragma unroll
  for (int i = 0; i < 2; ++i) {
    async_cp16(gA[i], As + rc[i] * LDT + sc[i]);
    async_cp16(gB[i], Bs + rc[i] * LDT + sc[i]);
  }
  __builtin_amdgcn_s_wait_asynccnt(0);
#else
  {
    v4i ra[2], rb[2];
#pragma unroll
    for (int i = 0; i < 2; ++i) { ra[i] = *(const v4i*)gA[i]; rb[i] = *(const v4i*)gB[i]; }
#pragma unroll
    for (int i = 0; i < 2; ++i) {
      *(v4i*)(As + rc[i] * LDT + sc[i]) = ra[i];
      *(v4i*)(Bs + rc[i] * LDT + sc[i]) = rb[i];
    }
  }
#endif
  __syncthreads();

  int nk = K / BK;
  for (int kb = 0; kb < nk; ++kb) {
    int cur = kb & 1, nxt = cur ^ 1;
    long koff = (long)(kb + 1) * BK;
    bool more = (kb + 1) < nk;
#if !HAS_ASYNC
    v4i ra[2], rb[2];
#endif
    if (more) {
#if HAS_ASYNC
#pragma unroll
      for (int i = 0; i < 2; ++i) {
        async_cp16(gA[i] + koff, As + nxt * (BM * LDT) + rc[i] * LDT + sc[i]);
        async_cp16(gB[i] + koff, Bs + nxt * (BN * LDT) + rc[i] * LDT + sc[i]);
      }
#else
#pragma unroll
      for (int i = 0; i < 2; ++i) {
        ra[i] = *(const v4i*)(gA[i] + koff);
        rb[i] = *(const v4i*)(gB[i] + koff);
      }
#endif
    }
    const bf16_t* Ac = As + cur * (BM * LDT);
    const bf16_t* Bc = Bs + cur * (BN * LDT);
    v16bf af[2], bfv[4];
#pragma unroll
    for (int i = 0; i < 2; ++i) af[i] = load_a_frag(Ac + (wm + i * 16) * LDT, lane, LDT);
#pragma unroll
    for (int j = 0; j < 4; ++j) bfv[j] = load_b_frag(Bc + (wn + j * 16) * LDT, lane, LDT);
#pragma unroll
    for (int i = 0; i < 2; ++i)
#pragma unroll
      for (int j = 0; j < 4; ++j)
        acc[i][j] = __builtin_amdgcn_wmma_f32_16x16x32_bf16(
            false, af[i], false, bfv[j], (short)0, acc[i][j], false, false);
    if (more) {
#if HAS_ASYNC
      __builtin_amdgcn_s_wait_asynccnt(0);
#else
#pragma unroll
      for (int i = 0; i < 2; ++i) {
        *(v4i*)(As + nxt * (BM * LDT) + rc[i] * LDT + sc[i]) = ra[i];
        *(v4i*)(Bs + nxt * (BN * LDT) + rc[i] * LDT + sc[i]) = rb[i];
      }
#endif
    }
    __syncthreads();
  }
  int n0 = lane & 15;
  int m0 = (lane >> 4) * 8;
#pragma unroll
  for (int i = 0; i < 2; ++i)
#pragma unroll
    for (int j = 0; j < 4; ++j) {
      float* cp = C + (long)(bm + wm + i * 16 + m0) * N + bn + wn + j * 16 + n0;
#pragma unroll
      for (int r = 0; r < 8; ++r) cp[(long)r * N] = acc[i][j][r];
    }
}

// ---------------------------------------------------------------- fused attention
// grid (S/64, NH, B), 128 threads (4 waves). Wave w owns q rows [qb*64+w*16, +16).
// K is [B,NKV,S,HD]; V is [B,NKV,HD,S] (pre-transposed) -> both tiles are direct copies.
__global__ __launch_bounds__(128) void attn_kernel(const bf16_t* __restrict__ Q,
                                                   const bf16_t* __restrict__ Kh,
                                                   const bf16_t* __restrict__ Vh,
                                                   const int* __restrict__ amask,
                                                   bf16_t* __restrict__ Out,
                                                   int sliding) {
  __shared__ bf16_t Qs[64 * 72];
  __shared__ bf16_t Ks[2 * 64 * 72];   // Ks[buf][kj][d]
  __shared__ bf16_t Vs[2 * 64 * 72];   // Vs[buf][d][kj]
  __shared__ bf16_t Ps[4 * 16 * 72];   // per-wave P tile
  int qb = blockIdx.x, h = blockIdx.y, b = blockIdx.z;
  int tid = threadIdx.x, wave = tid >> 5, lane = tid & 31;
  int qbase = qb * 64;
  int kvh = h >> 1;                    // GQA repeat = 2
  const bf16_t* Qp = Q + (((long)(b * NH_ + h)) * S_ + qbase) * HD_;
  const bf16_t* Kp = Kh + ((long)(b * NKV_ + kvh)) * S_ * HD_;
  const bf16_t* Vp = Vh + ((long)(b * NKV_ + kvh)) * HD_ * S_;
  const int* am = amask + (long)b * S_;

  int rk[4], sk[4];
#pragma unroll
  for (int i = 0; i < 4; ++i) {
    int c = tid + i * 128;             // 512 chunks per 64x64 tile
    rk[i] = c >> 3; sk[i] = (c & 7) * 8;
  }
  // stage Q (one-time)
#pragma unroll
  for (int i = 0; i < 4; ++i)
    *(v4i*)(Qs + rk[i] * 72 + sk[i]) = *(const v4i*)(Qp + (long)rk[i] * HD_ + sk[i]);
  __syncthreads();
  v16bf aq0 = load_a_frag(Qs + (wave * 16) * 72, lane, 72);
  v16bf aq1 = load_a_frag(Qs + (wave * 16) * 72 + 32, lane, 72);

  v8f Oacc[4];
  float mrow[8], lrow[8];
#pragma unroll
  for (int t = 0; t < 4; ++t) Oacc[t] = v8f_zero();
#pragma unroll
  for (int r = 0; r < 8; ++r) { mrow[r] = -3.0e38f; lrow[r] = 0.0f; }

  int kb0 = 0, kb1 = S_ / 64;
  if (sliding) {
    int lo = qbase - W_; if (lo < 0) lo = 0;
    int hi = qbase + 63 + W_; if (hi > S_ - 1) hi = S_ - 1;
    kb0 = lo >> 6; kb1 = (hi >> 6) + 1;
  }
  int n_local = lane & 15;
  int m_local = (lane >> 4) * 8;
  int qrow = qbase + wave * 16 + m_local;

  // prologue: stage k-block kb0 into buffer 0
#if HAS_ASYNC
#pragma unroll
  for (int i = 0; i < 4; ++i) {
    async_cp16(Kp + (long)(kb0 * 64 + rk[i]) * HD_ + sk[i], Ks + rk[i] * 72 + sk[i]);
    async_cp16(Vp + (long)rk[i] * S_ + kb0 * 64 + sk[i],    Vs + rk[i] * 72 + sk[i]);
  }
  __builtin_amdgcn_s_wait_asynccnt(0);
#else
  {
    v4i tk[4], tv[4];
#pragma unroll
    for (int i = 0; i < 4; ++i) {
      tk[i] = *(const v4i*)(Kp + (long)(kb0 * 64 + rk[i]) * HD_ + sk[i]);
      tv[i] = *(const v4i*)(Vp + (long)rk[i] * S_ + kb0 * 64 + sk[i]);
    }
#pragma unroll
    for (int i = 0; i < 4; ++i) {
      *(v4i*)(Ks + rk[i] * 72 + sk[i]) = tk[i];
      *(v4i*)(Vs + rk[i] * 72 + sk[i]) = tv[i];
    }
  }
#endif
  __syncthreads();

  for (int kb = kb0; kb < kb1; ++kb) {
    int cur = (kb - kb0) & 1, nxt = cur ^ 1;
    int kstart = kb * 64;
    bool more = (kb + 1) < kb1;
#if !HAS_ASYNC
    v4i tk[4], tv[4];
#endif
    if (more) {
      int ks2 = kstart + 64;
#if HAS_ASYNC
#pragma unroll
      for (int i = 0; i < 4; ++i) {
        async_cp16(Kp + (long)(ks2 + rk[i]) * HD_ + sk[i],
                   Ks + nxt * (64 * 72) + rk[i] * 72 + sk[i]);
        async_cp16(Vp + (long)rk[i] * S_ + ks2 + sk[i],
                   Vs + nxt * (64 * 72) + rk[i] * 72 + sk[i]);
      }
#else
#pragma unroll
      for (int i = 0; i < 4; ++i) {
        tk[i] = *(const v4i*)(Kp + (long)(ks2 + rk[i]) * HD_ + sk[i]);
        tv[i] = *(const v4i*)(Vp + (long)rk[i] * S_ + ks2 + sk[i]);
      }
#endif
    }
    const bf16_t* Kc = Ks + cur * (64 * 72);
    const bf16_t* Vc = Vs + cur * (64 * 72);

    // ---- scores = (Q K^T) * scale + mask
    v8f sf[4];
#pragma unroll
    for (int j = 0; j < 4; ++j) {
      v8f s = v8f_zero();
      v16bf bk0 = load_b_frag(Kc + (j * 16) * 72, lane, 72);
      v16bf bk1 = load_b_frag(Kc + (j * 16) * 72 + 32, lane, 72);
      s = __builtin_amdgcn_wmma_f32_16x16x32_bf16(false, aq0, false, bk0, (short)0, s, false, false);
      s = __builtin_amdgcn_wmma_f32_16x16x32_bf16(false, aq1, false, bk1, (short)0, s, false, false);
      int ncol = kstart + j * 16 + n_local;
      float pad = (am[ncol] == 0) ? NEG_ : 0.0f;
#pragma unroll
      for (int r = 0; r < 8; ++r) {
        float v = s[r] * 0.125f;
        float add = pad;
        if (sliding) {
          int dq = (qrow + r) - ncol;
          if (dq > W_ || dq < -W_) add = NEG_;
        }
        s[r] = v + add;
      }
      sf[j] = s;
    }
    // ---- online softmax
    float rmax[8];
#pragma unroll
    for (int r = 0; r < 8; ++r) {
      float v = fmaxf(fmaxf(sf[0][r], sf[1][r]), fmaxf(sf[2][r], sf[3][r]));
#pragma unroll
      for (int m = 8; m; m >>= 1) v = fmaxf(v, __shfl_xor(v, m, 32));
      rmax[r] = v;
    }
    float alpha[8];
#pragma unroll
    for (int r = 0; r < 8; ++r) {
      float mnew = fmaxf(mrow[r], rmax[r]);
      alpha[r] = __expf(mrow[r] - mnew);
      mrow[r] = mnew;
    }
#pragma unroll
    for (int j = 0; j < 4; ++j)
#pragma unroll
      for (int r = 0; r < 8; ++r) {
        float p = __expf(sf[j][r] - mrow[r]);
        sf[j][r] = p;
        Ps[(wave * 16 + m_local + r) * 72 + j * 16 + n_local] = (bf16_t)p;
      }
#pragma unroll
    for (int r = 0; r < 8; ++r) {
      float v = sf[0][r] + sf[1][r] + sf[2][r] + sf[3][r];
#pragma unroll
      for (int m = 8; m; m >>= 1) v += __shfl_xor(v, m, 32);
      lrow[r] = lrow[r] * alpha[r] + v;
    }
#pragma unroll
    for (int t = 0; t < 4; ++t)
#pragma unroll
      for (int r = 0; r < 8; ++r) Oacc[t][r] = Oacc[t][r] * alpha[r];

    // ---- O += P x V  (same-wave LDS ops are in-order)
    v16bf ap0 = load_a_frag(Ps + (wave * 16) * 72, lane, 72);
    v16bf ap1 = load_a_frag(Ps + (wave * 16) * 72 + 32, lane, 72);
#pragma unroll
    for (int t = 0; t < 4; ++t) {
      v16bf bv0 = load_b_frag(Vc + (t * 16) * 72, lane, 72);
      v16bf bv1 = load_b_frag(Vc + (t * 16) * 72 + 32, lane, 72);
      Oacc[t] = __builtin_amdgcn_wmma_f32_16x16x32_bf16(false, ap0, false, bv0, (short)0, Oacc[t], false, false);
      Oacc[t] = __builtin_amdgcn_wmma_f32_16x16x32_bf16(false, ap1, false, bv1, (short)0, Oacc[t], false, false);
    }
    if (more) {
#if HAS_ASYNC
      __builtin_amdgcn_s_wait_asynccnt(0);
#else
#pragma unroll
      for (int i = 0; i < 4; ++i) {
        *(v4i*)(Ks + nxt * (64 * 72) + rk[i] * 72 + sk[i]) = tk[i];
        *(v4i*)(Vs + nxt * (64 * 72) + rk[i] * 72 + sk[i]) = tv[i];
      }
#endif
    }
    __syncthreads();
  }
  // ---- write out [B, S, NH*HD] bf16
#pragma unroll
  for (int t = 0; t < 4; ++t)
#pragma unroll
    for (int r = 0; r < 8; ++r) {
      int q = qbase + wave * 16 + m_local + r;
      float o = Oacc[t][r] / lrow[r];
      Out[((long)(b * S_ + q)) * (NH_ * HD_) + h * HD_ + t * 16 + n_local] = (bf16_t)o;
    }
}

// ---------------------------------------------------------------- host orchestration
extern "C" void kernel_launch(void* const* d_in, const int* in_sizes, int n_in,
                              void* d_out, int out_size, void* d_ws, size_t ws_size,
                              hipStream_t stream) {
  const int*   ids   = (const int*)d_in[0];
  const int*   amask = (const int*)d_in[1];
  const float* emb   = (const float*)d_in[2];
  const float* wq    = (const float*)d_in[3];
  const float* wk    = (const float*)d_in[4];
  const float* wv    = (const float*)d_in[5];
  const float* wo    = (const float*)d_in[6];
  const float* qnw   = (const float*)d_in[7];
  const float* knw   = (const float*)d_in[8];
  const float* ln1   = (const float*)d_in[9];
  const float* ln2   = (const float*)d_in[10];
  const float* wg    = (const float*)d_in[11];
  const float* wu    = (const float*)d_in[12];
  const float* wd    = (const float*)d_in[13];
  const float* fnorm = (const float*)d_in[14];

  char* ws = (char*)d_ws;
  size_t off = 0;
  auto alloc = [&](size_t bytes) -> char* {
    char* p = ws + off;
    off += (bytes + 255) & ~(size_t)255;
    return p;
  };
  const long T = (long)B_ * S_;   // 4096 tokens

  bf16_t* wq_b = (bf16_t*)alloc((size_t)L_ * H_ * (NH_ * HD_) * 2);   // [N,K] per layer
  bf16_t* wk_b = (bf16_t*)alloc((size_t)L_ * H_ * (NKV_ * HD_) * 2);
  bf16_t* wv_b = (bf16_t*)alloc((size_t)L_ * H_ * (NKV_ * HD_) * 2);
  bf16_t* wo_b = (bf16_t*)alloc((size_t)L_ * (NH_ * HD_) * H_ * 2);
  bf16_t* wg_b = (bf16_t*)alloc((size_t)L_ * H_ * F_ * 2);
  bf16_t* wu_b = (bf16_t*)alloc((size_t)L_ * H_ * F_ * 2);
  bf16_t* wd_b = (bf16_t*)alloc((size_t)L_ * F_ * H_ * 2);
  float*  h    = (float*)alloc((size_t)T * H_ * 4);
  bf16_t* xb   = (bf16_t*)alloc((size_t)T * H_ * 2);
  float*  qf   = (float*)alloc((size_t)T * (NH_ * HD_) * 4);
  float*  kf   = (float*)alloc((size_t)T * (NKV_ * HD_) * 4);
  float*  vf   = (float*)alloc((size_t)T * (NKV_ * HD_) * 4);
  bf16_t* Qh   = (bf16_t*)alloc((size_t)T * NH_ * HD_ * 2);
  bf16_t* Kh   = (bf16_t*)alloc((size_t)T * NKV_ * HD_ * 2);
  bf16_t* Vh   = (bf16_t*)alloc((size_t)T * NKV_ * HD_ * 2);
  bf16_t* atb  = (bf16_t*)alloc((size_t)T * (NH_ * HD_) * 2);
  float*  gf   = (float*)alloc((size_t)T * F_ * 4);
  float*  uf   = (float*)alloc((size_t)T * F_ * 4);
  bf16_t* mb   = (bf16_t*)alloc((size_t)T * F_ * 2);
  (void)ws_size; (void)in_sizes; (void)n_in; (void)out_size;

  auto cvtT = [&](const float* src, bf16_t* dst, int K, int N) {
    cvt_transpose<<<dim3(N / 32, K / 32, L_), dim3(32, 8), 0, stream>>>(src, dst, K, N);
  };
  cvtT(wq, wq_b, H_, NH_ * HD_);
  cvtT(wk, wk_b, H_, NKV_ * HD_);
  cvtT(wv, wv_b, H_, NKV_ * HD_);
  cvtT(wo, wo_b, NH_ * HD_, H_);
  cvtT(wg, wg_b, H_, F_);
  cvtT(wu, wu_b, H_, F_);
  cvtT(wd, wd_b, F_, H_);

  embed_gather<<<(int)T, 256, 0, stream>>>(ids, emb, h);

  for (int l = 0; l < L_; ++l) {
    rmsnorm_kernel<<<(int)T, 256, 0, stream>>>(h, ln1 + (long)l * H_, xb, nullptr);

    gemm_bf16<<<dim3((NH_ * HD_) / BN, (int)(T / BM)), 256, 0, stream>>>(
        xb, wq_b + (long)l * H_ * NH_ * HD_, qf, (int)T, NH_ * HD_, H_);
    gemm_bf16<<<dim3((NKV_ * HD_) / BN, (int)(T / BM)), 256, 0, stream>>>(
        xb, wk_b + (long)l * H_ * NKV_ * HD_, kf, (int)T, NKV_ * HD_, H_);
    gemm_bf16<<<dim3((NKV_ * HD_) / BN, (int)(T / BM)), 256, 0, stream>>>(
        xb, wv_b + (long)l * H_ * NKV_ * HD_, vf, (int)T, NKV_ * HD_, H_);

    qk_post<<<dim3(S_, NH_, B_), 32, 0, stream>>>(qf, qnw + (long)l * HD_, Qh, NH_);
    qk_post<<<dim3(S_, NKV_, B_), 32, 0, stream>>>(kf, knw + (long)l * HD_, Kh, NKV_);
    long nv = T * NKV_ * HD_;
    v_pack<<<(int)((nv + 255) / 256), 256, 0, stream>>>(vf, Vh);

    attn_kernel<<<dim3(S_ / 64, NH_, B_), 128, 0, stream>>>(Qh, Kh, Vh, amask, atb, l & 1);

    gemm_bf16<<<dim3(H_ / BN, (int)(T / BM)), 256, 0, stream>>>(
        atb, wo_b + (long)l * NH_ * HD_ * H_, qf, (int)T, H_, NH_ * HD_);
    add_res<<<(int)((T * H_ + 255) / 256), 256, 0, stream>>>(h, qf, T * H_);

    rmsnorm_kernel<<<(int)T, 256, 0, stream>>>(h, ln2 + (long)l * H_, xb, nullptr);
    gemm_bf16<<<dim3(F_ / BN, (int)(T / BM)), 256, 0, stream>>>(
        xb, wg_b + (long)l * H_ * F_, gf, (int)T, F_, H_);
    gemm_bf16<<<dim3(F_ / BN, (int)(T / BM)), 256, 0, stream>>>(
        xb, wu_b + (long)l * H_ * F_, uf, (int)T, F_, H_);
    silu_mul<<<(int)((T * F_ + 255) / 256), 256, 0, stream>>>(gf, uf, mb, T * F_);
    gemm_bf16<<<dim3(H_ / BN, (int)(T / BM)), 256, 0, stream>>>(
        mb, wd_b + (long)l * F_ * H_, uf, (int)T, H_, F_);
    add_res<<<(int)((T * H_ + 255) / 256), 256, 0, stream>>>(h, uf, T * H_);
  }

  rmsnorm_kernel<<<(int)T, 256, 0, stream>>>(h, fnorm, nullptr, (float*)d_out);
}
